// PerceiverAttention_67525475828179
// MI455X (gfx1250) — matrix operand order repaired and checked by reading, not compiled, not run on hardware
//
#include <hip/hip_runtime.h>
#include <hip/hip_bf16.h>

#define DIM 1024
#define HEADS 16
#define DIM_HEAD 64
#define INNER 1024
#define EPS_LN 1e-5f

typedef _Float16 half_t;
typedef __attribute__((ext_vector_type(4)))  _Float16 v4h;
typedef __attribute__((ext_vector_type(8)))  _Float16 v8h;
typedef __attribute__((ext_vector_type(16))) _Float16 v16h;
typedef __attribute__((ext_vector_type(8)))  float    v8f;
typedef __attribute__((ext_vector_type(4)))  float    v4f;

__device__ __forceinline__ v16h cat8(v8h lo, v8h hi) {
  return __builtin_shufflevector(lo, hi, 0, 1, 2, 3, 4, 5, 6, 7, 8, 9, 10, 11, 12, 13, 14, 15);
}

__device__ __forceinline__ v8f wmma_f16(v16h a, v16h b, v8f c) {
  // D = A(16x32 f16) * B(32x16 f16) + C(16x16 f32)
  return __builtin_amdgcn_wmma_f32_16x16x32_f16(false, a, false, b, (short)0, c, false, false);
}

// Async DMA: per-lane 16B global -> LDS, tracked by ASYNCcnt (no VGPR staging).
// LDS address = low 32 bits of the flat shared pointer (LDS aperture rule).
__device__ __forceinline__ void async_ld_b128(const half_t* gsrc, half_t* ldst) {
  uint32_t laddr = (uint32_t)(uintptr_t)ldst;
  uint64_t gaddr = (uint64_t)(uintptr_t)gsrc;
  asm volatile("global_load_async_to_lds_b128 %0, %1, off"
               :: "v"(laddr), "v"(gaddr)
               : "memory");
}

__device__ __forceinline__ void wait_async_le4() {
  asm volatile("s_wait_asynccnt 0x4" ::: "memory");
}
__device__ __forceinline__ void wait_async_0() {
  asm volatile("s_wait_asynccnt 0x0" ::: "memory");
}

// ---------------------------------------------------------------------------
// Fused LayerNorm (row of 1024 fp32) -> f16, one block (8 waves) per row.
// ---------------------------------------------------------------------------
__global__ __launch_bounds__(256) void ln_cast_kernel(
    const float* __restrict__ x, const float* __restrict__ g, const float* __restrict__ b,
    half_t* __restrict__ y)
{
  const int row  = blockIdx.x;
  const int tid  = threadIdx.x;
  const int lane = tid & 31, wave = tid >> 5;
  const float* xr = x + (size_t)row * DIM;
  v4f v = *(const v4f*)(xr + tid * 4);
  float s  = v[0] + v[1] + v[2] + v[3];
  float s2 = v[0] * v[0] + v[1] * v[1] + v[2] * v[2] + v[3] * v[3];
#pragma unroll
  for (int off = 16; off > 0; off >>= 1) {
    s  += __shfl_xor(s, off, 32);
    s2 += __shfl_xor(s2, off, 32);
  }
  __shared__ float red[2][8];
  if (lane == 0) { red[0][wave] = s; red[1][wave] = s2; }
  __syncthreads();
  float ts = 0.f, ts2 = 0.f;
#pragma unroll
  for (int w = 0; w < 8; ++w) { ts += red[0][w]; ts2 += red[1][w]; }
  const float mu  = ts * (1.0f / DIM);
  const float var = ts2 * (1.0f / DIM) - mu * mu;
  const float rs  = rsqrtf(var + EPS_LN);
  v4f gg = *(const v4f*)(g + tid * 4);
  v4f bb = *(const v4f*)(b + tid * 4);
  v4h o;
#pragma unroll
  for (int e = 0; e < 4; ++e) o[e] = (half_t)((v[e] - mu) * rs * gg[e] + bb[e]);
  *(v4h*)(y + (size_t)row * DIM + tid * 4) = o;
}

// ---------------------------------------------------------------------------
// fp32 -> f16 weight downconvert (optionally folding a scalar scale).
// ---------------------------------------------------------------------------
__global__ __launch_bounds__(256) void cvt_f16_kernel(const float* __restrict__ src,
                                                      half_t* __restrict__ dst,
                                                      float scale, int n4)
{
  int i = blockIdx.x * blockDim.x + threadIdx.x;
  if (i >= n4) return;
  v4f v = *(const v4f*)(src + (size_t)i * 4);
  v4h o;
#pragma unroll
  for (int e = 0; e < 4; ++e) o[e] = (half_t)(v[e] * scale);
  *(v4h*)(dst + (size_t)i * 4) = o;
}

// ---------------------------------------------------------------------------
// C[rows,N] = A[rows,K] @ W[N,K]^T via v_wmma_f32_16x16x32_f16.
// Block: 256 threads (8 waves), tile 128x128, k-step 32, DOUBLE-BUFFERED
// via global_load_async_to_lds_b128: DMA for tile t+1 overlaps WMMA on tile t.
// Wave (wm,wn) in 2x4 owns a 64x32 sub-tile -> 8 accumulators.
// ---------------------------------------------------------------------------
template <bool OUT_F32>
__global__ __launch_bounds__(256) void gemm_wmma_kernel(
    const half_t* __restrict__ A,   // (rows, K) row-major
    const half_t* __restrict__ W,   // (N, K) row-major
    half_t* __restrict__ Ch, float* __restrict__ Cf,
    int K, int N)
{
  const int LDT = 40;  // 32 + 8 halves pad: 80B rows (16B-aligned, bank-staggered)
  __shared__ __attribute__((aligned(16))) half_t As[2][128][LDT];
  __shared__ __attribute__((aligned(16))) half_t Bs[2][128][LDT];
  const int tid  = threadIdx.x;
  const int wave = tid >> 5, lane = tid & 31;
  const int hi   = lane >> 4, ln = lane & 15;
  const int wm   = wave >> 2, wn = wave & 3;
  const size_t bm = (size_t)blockIdx.x * 128;
  const size_t bn = (size_t)blockIdx.y * 128;

  // Cooperative async tile load: 512 b128 slots per tile pair of buffers.
  const int r0 = tid >> 2;            // rows 0..63
  const int r1 = (tid + 256) >> 2;    // rows 64..127
  const int sg = (tid & 3) * 8;       // k-segment within tile
  auto issue_tile = [&](int k0, int buf) {
    async_ld_b128(A + (bm + r0) * (size_t)K + k0 + sg, &As[buf][r0][sg]);
    async_ld_b128(A + (bm + r1) * (size_t)K + k0 + sg, &As[buf][r1][sg]);
    async_ld_b128(W + (bn + r0) * (size_t)K + k0 + sg, &Bs[buf][r0][sg]);
    async_ld_b128(W + (bn + r1) * (size_t)K + k0 + sg, &Bs[buf][r1][sg]);
  };

  const v8f vzero = {};
  v8f acc[4][2];
#pragma unroll
  for (int i = 0; i < 4; ++i)
#pragma unroll
    for (int j = 0; j < 2; ++j) acc[i][j] = vzero;

  issue_tile(0, 0);
  const int T = K / 32;
  for (int t = 0; t < T; ++t) {
    const int buf = t & 1;
    if (t + 1 < T) {
      issue_tile((t + 1) * 32, buf ^ 1);  // 8 outstanding async ops now
      wait_async_le4();                   // tile t resident; t+1 still in flight
    } else {
      wait_async_0();
    }
    __syncthreads();
    v16h af[4], bf[2];
#pragma unroll
    for (int i = 0; i < 4; ++i) {
      const half_t* p = &As[buf][wm * 64 + i * 16 + ln][0];
      af[i] = cat8(*(const v8h*)(p + hi * 8), *(const v8h*)(p + 16 + hi * 8));
    }
#pragma unroll
    for (int j = 0; j < 2; ++j) {
      const half_t* p = &Bs[buf][wn * 32 + j * 16 + ln][0];
      bf[j] = cat8(*(const v8h*)(p + hi * 8), *(const v8h*)(p + 16 + hi * 8));
    }
#pragma unroll
    for (int i = 0; i < 4; ++i)
#pragma unroll
      for (int j = 0; j < 2; ++j) acc[i][j] = wmma_f16(af[i], bf[j], acc[i][j]);
    __syncthreads();  // all waves done reading buf before iter t+1 overwrites it
  }

  // C/D layout: lane holds col n = lane&15; component j -> row hi*8 + j.
#pragma unroll
  for (int i = 0; i < 4; ++i)
#pragma unroll
    for (int j = 0; j < 2; ++j) {
      size_t m = bm + wm * 64 + i * 16 + hi * 8;
      size_t n = bn + wn * 32 + j * 16 + ln;
#pragma unroll
      for (int c = 0; c < 8; ++c) {
        if (OUT_F32) Cf[(m + c) * (size_t)N + n] = acc[i][j][c];
        else         Ch[(m + c) * (size_t)N + n] = (half_t)acc[i][j][c];
      }
    }
}

// ---------------------------------------------------------------------------
// Flash attention, one block per (batch, head). 8 waves; wave w owns q rows
// [16w,16w+16). Q frags persistent in registers; K tile via async-to-LDS,
// V^T scatter on the sync path; online softmax; P staged in wave-private LDS
// rows for the P@V WMMA.
// ---------------------------------------------------------------------------
__global__ __launch_bounds__(256) void attn_kernel(
    const half_t* __restrict__ q,   // (B, 128, INNER), scale pre-folded
    const half_t* __restrict__ kv,  // (B, S, 2*INNER): [k | v]
    half_t* __restrict__ o,         // (B, 128, INNER)
    int S)
{
  const int SB  = 64;
  const int LKT = DIM_HEAD + 8;  // 72 halves = 144B rows
  const int LVT = SB + 8;        // 72
  __shared__ __attribute__((aligned(16))) half_t Ks[SB][LKT];        // K tile [s][d]
  __shared__ __attribute__((aligned(16))) half_t Vt[DIM_HEAD][LVT];  // V^T    [d][s]
  __shared__ __attribute__((aligned(16))) half_t Ps[128][LVT];       // softmax probs
  const int b = blockIdx.x, h = blockIdx.y;
  const int tid  = threadIdx.x;
  const int wave = tid >> 5, lane = tid & 31;
  const int hi   = lane >> 4, ln = lane & 15;

  // Persistent A-fragments of q (16 rows x 64 K per wave -> 2 k-steps).
  const half_t* qrow = q + ((size_t)b * 128 + wave * 16 + ln) * (size_t)INNER + h * DIM_HEAD;
  v16h qf[2];
#pragma unroll
  for (int ks = 0; ks < 2; ++ks)
    qf[ks] = cat8(*(const v8h*)(qrow + ks * 32 + hi * 8),
                  *(const v8h*)(qrow + ks * 32 + 16 + hi * 8));

  float mrow[8], lrow[8];
#pragma unroll
  for (int j = 0; j < 8; ++j) { mrow[j] = -3.0e38f; lrow[j] = 0.0f; }
  const v8f vzero = {};
  v8f oacc[4];
#pragma unroll
  for (int i = 0; i < 4; ++i) oacc[i] = vzero;

  const size_t kvb = (size_t)b * (size_t)S * (size_t)(2 * INNER);
  for (int s0 = 0; s0 < S; s0 += SB) {
    // K tile (64 x 64 halves): async DMA straight into LDS, 2 ops per thread.
    // V tile: sync load + transposed scatter (2B granularity needs VGPR path).
#pragma unroll
    for (int it = 0; it < 2; ++it) {
      int slot = tid + it * 256;
      int r = slot >> 3, seg = (slot & 7) * 8;
      const half_t* src = kv + kvb + (size_t)(s0 + r) * (2 * INNER) + h * DIM_HEAD + seg;
      async_ld_b128(src, &Ks[r][seg]);
      v8h vv = *(const v8h*)(src + INNER);
#pragma unroll
      for (int e = 0; e < 8; ++e) Vt[seg + e][r] = vv[e];
    }
    wait_async_0();
    __syncthreads();

    // sim = q @ k^T : wave tile 16 x 64 (4 accumulators), K = 64.
    v8f sim[4];
#pragma unroll
    for (int j16 = 0; j16 < 4; ++j16) {
      v8f c = vzero;
#pragma unroll
      for (int ks = 0; ks < 2; ++ks) {
        const half_t* p = &Ks[j16 * 16 + ln][ks * 32];
        c = wmma_f16(qf[ks], cat8(*(const v8h*)(p + hi * 8), *(const v8h*)(p + 16 + hi * 8)), c);
      }
      sim[j16] = c;
    }

    // Row max over 64 keys: componentwise across chunks, then 16-lane xor tree.
    v8f cmax = sim[0];
#pragma unroll
    for (int j16 = 1; j16 < 4; ++j16)
#pragma unroll
      for (int j = 0; j < 8; ++j) cmax[j] = fmaxf(cmax[j], sim[j16][j]);
#pragma unroll
    for (int off = 1; off < 16; off <<= 1)
#pragma unroll
      for (int j = 0; j < 8; ++j) cmax[j] = fmaxf(cmax[j], __shfl_xor(cmax[j], off, 32));

    float rescale[8];
#pragma unroll
    for (int j = 0; j < 8; ++j) {
      float mn = fmaxf(mrow[j], cmax[j]);
      rescale[j] = __expf(mrow[j] - mn);
      mrow[j] = mn;
      lrow[j] *= rescale[j];
    }
#pragma unroll
    for (int i = 0; i < 4; ++i)
#pragma unroll
      for (int j = 0; j < 8; ++j) oacc[i][j] *= rescale[j];

    // p = exp(sim - m); accumulate l; stage p to wave-private LDS rows.
    v8f csum = vzero;
#pragma unroll
    for (int j16 = 0; j16 < 4; ++j16)
#pragma unroll
      for (int j = 0; j < 8; ++j) {
        float pv = __expf(sim[j16][j] - mrow[j]);
        csum[j] += pv;
        Ps[wave * 16 + hi * 8 + j][j16 * 16 + ln] = (half_t)pv;
      }
#pragma unroll
    for (int off = 1; off < 16; off <<= 1)
#pragma unroll
      for (int j = 0; j < 8; ++j) csum[j] += __shfl_xor(csum[j], off, 32);
#pragma unroll
    for (int j = 0; j < 8; ++j) lrow[j] += csum[j];

    // out += p @ v : A rows wave-private (same-wave LDS ordering), B = V^T.
#pragma unroll
    for (int n16 = 0; n16 < 4; ++n16) {
      v8f c = oacc[n16];
#pragma unroll
      for (int ks = 0; ks < 2; ++ks) {
        const half_t* pa = &Ps[wave * 16 + ln][ks * 32];
        const half_t* pb = &Vt[n16 * 16 + ln][ks * 32];
        c = wmma_f16(cat8(*(const v8h*)(pa + hi * 8),  *(const v8h*)(pa + 16 + hi * 8)),
                     cat8(*(const v8h*)(pb + hi * 8), *(const v8h*)(pb + 16 + hi * 8)), c);
      }
      oacc[n16] = c;
    }
    __syncthreads();
  }

  // Normalize and store to (B, M, H*DH) = "b h n d -> b n (h d)".
#pragma unroll
  for (int n16 = 0; n16 < 4; ++n16)
#pragma unroll
    for (int j = 0; j < 8; ++j) {
      size_t m = (size_t)(wave * 16 + hi * 8 + j);
      o[((size_t)b * 128 + m) * INNER + h * DIM_HEAD + n16 * 16 + ln] =
          (half_t)(oacc[n16][j] / lrow[j]);
    }
}

// ---------------------------------------------------------------------------
// Host-side orchestration (all launches on `stream`; scratch carved from ws).
// ---------------------------------------------------------------------------
extern "C" void kernel_launch(void* const* d_in, const int* in_sizes, int n_in,
                              void* d_out, int out_size, void* d_ws, size_t ws_size,
                              hipStream_t stream) {
  (void)in_sizes; (void)n_in; (void)out_size; (void)ws_size;
  const int B = 16, S = 2048, M = 128;
  const float* x   = (const float*)d_in[0];
  const float* lat = (const float*)d_in[1];
  const float* gx  = (const float*)d_in[2];
  const float* bx  = (const float*)d_in[3];
  const float* gl  = (const float*)d_in[4];
  const float* bl  = (const float*)d_in[5];
  const float* Wq  = (const float*)d_in[6];
  const float* Wkv = (const float*)d_in[7];
  const float* Wo  = (const float*)d_in[8];
  float* out = (float*)d_out;

  char* ws = (char*)d_ws;
  size_t off = 0;
  auto carve = [&](size_t bytes) -> half_t* {
    half_t* p = (half_t*)(ws + off);
    off += (bytes + 255) & ~(size_t)255;
    return p;
  };
  half_t* xn   = carve((size_t)B * S * DIM * 2);        //  64 MB
  half_t* lnl  = carve((size_t)B * M * DIM * 2);        //   4 MB
  half_t* Wqh  = carve((size_t)INNER * DIM * 2);        //   2 MB
  half_t* Wkvh = carve((size_t)2 * INNER * DIM * 2);    //   4 MB
  half_t* Woh  = carve((size_t)DIM * INNER * 2);        //   2 MB
  half_t* qh   = carve((size_t)B * M * INNER * 2);      //   4 MB
  half_t* kvh  = carve((size_t)B * S * 2 * INNER * 2);  // 128 MB
  half_t* oh   = carve((size_t)B * M * INNER * 2);      //   4 MB

  ln_cast_kernel<<<B * S, 256, 0, stream>>>(x, gx, bx, xn);
  ln_cast_kernel<<<B * M, 256, 0, stream>>>(lat, gl, bl, lnl);

  const float qscale = 0.125f;  // DIM_HEAD^-0.5, folded into Wq
  cvt_f16_kernel<<<(INNER * DIM / 4 + 255) / 256, 256, 0, stream>>>(Wq, Wqh, qscale, INNER * DIM / 4);
  cvt_f16_kernel<<<(2 * INNER * DIM / 4 + 255) / 256, 256, 0, stream>>>(Wkv, Wkvh, 1.0f, 2 * INNER * DIM / 4);
  cvt_f16_kernel<<<(DIM * INNER / 4 + 255) / 256, 256, 0, stream>>>(Wo, Woh, 1.0f, DIM * INNER / 4);

  // q = ln(latents) @ (scale*Wq)^T            (2048 x 1024 x 1024)
  gemm_wmma_kernel<false><<<dim3(B * M / 128, INNER / 128), 256, 0, stream>>>(
      lnl, Wqh, qh, nullptr, DIM, INNER);
  // kv = ln(x) @ Wkv^T                        (32768 x 2048 x 1024)
  gemm_wmma_kernel<false><<<dim3(B * S / 128, 2 * INNER / 128), 256, 0, stream>>>(
      xn, Wkvh, kvh, nullptr, DIM, 2 * INNER);
  // flash attention per (b, h)
  attn_kernel<<<dim3(B, HEADS), 256, 0, stream>>>(qh, kvh, oh, S);
  // out = attn_out @ Wo^T -> fp32             (2048 x 1024 x 1024)
  gemm_wmma_kernel<true><<<dim3(B * M / 128, DIM / 128), 256, 0, stream>>>(
      oh, Woh, nullptr, out, INNER, DIM);
}
